// LocationPredictor_32753420599614
// MI455X (gfx1250) — matrix operand order repaired and checked by reading, not compiled, not run on hardware
//
#include <hip/hip_runtime.h>

#define BB   128
#define TT   256
#define EE   512
#define HDM  512
#define HSZ  1024
#define NG   2048      // 4*HD gates
#define KCAT 1024      // E + HD

typedef __attribute__((ext_vector_type(16))) __bf16 v16bf;
typedef __attribute__((ext_vector_type(8)))  float  v8f;

union Frag { v16bf v; uint4 q[2]; };

__device__ __forceinline__ unsigned short f2bf(float f) {
    unsigned int u = __float_as_uint(f);
    u += 0x7FFFu + ((u >> 16) & 1u);            // round-to-nearest-even
    return (unsigned short)(u >> 16);
}
__device__ __forceinline__ float bf2f(unsigned short h) {
    return __uint_as_float(((unsigned int)h) << 16);
}
__device__ __forceinline__ float sigf(float x) { return 1.0f / (1.0f + __expf(-x)); }

__device__ __forceinline__ void load_fragA(Frag& f, const unsigned short* row, int kh) {
    f.q[0] = *(const uint4*)(row + kh * 8);
    f.q[1] = *(const uint4*)(row + 16 + kh * 8);
}
__device__ __forceinline__ void load_fragB(Frag& f, const unsigned short* row, int kh) {
    f.q[0] = *(const uint4*)(row + kh * 8);
    f.q[1] = *(const uint4*)(row + 16 + kh * 8);
}
#define WMMA_BF16(A, B, C) \
    __builtin_amdgcn_wmma_f32_16x16x32_bf16(false, (A).v, false, (B).v, (short)0, (C), false, false)

// ---------------------------------------------------------------- prep kernels
__global__ void k_f32_to_bf16(const float* __restrict__ in, unsigned short* __restrict__ out, int n) {
    int i = blockIdx.x * blockDim.x + threadIdx.x;
    if (i < n) out[i] = f2bf(in[i]);
}

// Wcat[dir][n][k] : k<512 -> W_ih[n][k], else W_hh[n][k-512]   (B^T layout, N x K)
__global__ void k_build_wcat(const float* __restrict__ Wihf, const float* __restrict__ Whhf,
                             const float* __restrict__ Wihb, const float* __restrict__ Whhb,
                             unsigned short* __restrict__ Wcat) {
    int i = blockIdx.x * blockDim.x + threadIdx.x;          // < 2*2048*1024
    int dir = i >> 21;
    int rem = i & ((1 << 21) - 1);
    int n = rem >> 10, k = rem & 1023;
    const float* Wih = dir ? Wihb : Wihf;
    const float* Whh = dir ? Whhb : Whhf;
    float v = (k < EE) ? Wih[n * EE + k] : Whh[n * HDM + (k - EE)];
    Wcat[i] = f2bf(v);
}

// conv weights -> (co, tap*1024+ci) bf16, taps = 4 cross positions
__global__ void k_build_convw(const float* __restrict__ conv_w, unsigned short* __restrict__ out) {
    int i = blockIdx.x * blockDim.x + threadIdx.x;          // < 1024*4096
    int co = i >> 12;
    int rem = i & 4095;
    int d = rem >> 10, ci = rem & 1023;
    const int kyv[4] = {0, 1, 1, 2};
    const int kxv[4] = {1, 0, 2, 1};
    float v = conv_w[((co * HSZ + ci) * 3 + kyv[d]) * 3 + kxv[d]];
    out[i] = f2bf(v);
}

__global__ void k_init_state(float* __restrict__ c_state, unsigned short* __restrict__ hbf) {
    int i = blockIdx.x * blockDim.x + threadIdx.x;          // < 2*128*512
    c_state[i] = 0.0f;
    hbf[i] = 0;
}

__global__ void k_ctrl_init(const float* __restrict__ feat_ctrl, float* __restrict__ ctrl,
                            unsigned short* __restrict__ catbuf) {
    int i = blockIdx.x * blockDim.x + threadIdx.x;          // < 128*1024
    int b = i >> 10, h = i & 1023;
    float v = feat_ctrl[h];
    ctrl[i] = v;
    catbuf[b * 2048 + HSZ + h] = f2bf(v);
}

// ------------------------------------------------- LSTM step: gates = [x_t|h] @ Wcat^T + bias
// One wave computes a 32x32 output tile (2x2 WMMA register tiling, A/B fragment reuse)
__global__ void k_lstm_step_gemm(const int* __restrict__ Xs, const unsigned short* __restrict__ embbf,
                                 const unsigned short* __restrict__ hbf,
                                 const unsigned short* __restrict__ Wcat,
                                 const float* __restrict__ bias_f, const float* __restrict__ bias_b,
                                 float* __restrict__ gates, int step) {
    int lane = threadIdx.x;                       // wave32
    int n0 = blockIdx.x * 32;
    int m0 = blockIdx.y * 32;
    int dir = blockIdx.z;
    int t = dir ? (TT - 1 - step) : step;
    int i  = lane & 15;
    int kh = lane >> 4;

    int b0 = m0 + i, b1 = m0 + 16 + i;
    const unsigned short* xrow0 = embbf + (size_t)Xs[b0 * TT + t] * EE;
    const unsigned short* xrow1 = embbf + (size_t)Xs[b1 * TT + t] * EE;
    const unsigned short* hrow0 = hbf + ((size_t)dir * BB + b0) * HDM;
    const unsigned short* hrow1 = hbf + ((size_t)dir * BB + b1) * HDM;
    const unsigned short* wrow0 = Wcat + ((size_t)dir * NG + n0 + i) * KCAT;
    const unsigned short* wrow1 = wrow0 + (size_t)16 * KCAT;

    v8f acc00 = {}, acc01 = {}, acc10 = {}, acc11 = {};
    for (int kb = 0; kb < KCAT; kb += 32) {
        const unsigned short* a0 = (kb < EE) ? (xrow0 + kb) : (hrow0 + (kb - EE));
        const unsigned short* a1 = (kb < EE) ? (xrow1 + kb) : (hrow1 + (kb - EE));
        Frag fa0, fa1, fb0, fb1;
        load_fragA(fa0, a0, kh);
        load_fragA(fa1, a1, kh);
        load_fragB(fb0, wrow0 + kb, kh);
        load_fragB(fb1, wrow1 + kb, kh);
        __builtin_prefetch(wrow0 + kb + 64, 0, 1);   // global_prefetch_b8
        acc00 = WMMA_BF16(fa0, fb0, acc00);
        acc01 = WMMA_BF16(fa0, fb1, acc01);
        acc10 = WMMA_BF16(fa1, fb0, acc10);
        acc11 = WMMA_BF16(fa1, fb1, acc11);
    }
    int r0 = m0 + kh * 8, r1 = m0 + 16 + kh * 8;
    int c0 = n0 + i,      c1 = n0 + 16 + i;
    const float* bias = dir ? bias_b : bias_f;
    float bv0 = bias[c0], bv1 = bias[c1];
    for (int v = 0; v < 8; ++v) {
        gates[((size_t)dir * BB + r0 + v) * NG + c0] = acc00[v] + bv0;
        gates[((size_t)dir * BB + r0 + v) * NG + c1] = acc01[v] + bv1;
        gates[((size_t)dir * BB + r1 + v) * NG + c0] = acc10[v] + bv0;
        gates[((size_t)dir * BB + r1 + v) * NG + c1] = acc11[v] + bv1;
    }
}

__global__ void k_lstm_cell(const float* __restrict__ gates, float* __restrict__ c_state,
                            unsigned short* __restrict__ hbf, unsigned short* __restrict__ hidden,
                            int step) {
    int idx = blockIdx.x * blockDim.x + threadIdx.x;   // < 2*128*512
    int dir = idx >> 16;
    int rem = idx & 65535;
    int b = rem >> 9, j = rem & 511;
    const float* g = gates + ((size_t)dir * BB + b) * NG;
    float i_ = sigf(g[j]);
    float f_ = sigf(g[HDM + j]);
    float gg = tanhf(g[2 * HDM + j]);
    float o_ = sigf(g[3 * HDM + j]);
    size_t ci = ((size_t)dir * BB + b) * HDM + j;
    float c = f_ * c_state[ci] + i_ * gg;
    c_state[ci] = c;
    float h = o_ * tanhf(c);
    hbf[ci] = f2bf(h);
    int t = dir ? (TT - 1 - step) : step;
    hidden[((size_t)b * TT + t) * HSZ + dir * HDM + j] = f2bf(h);
}

// ------------------------------------------------- generic bf16 WMMA GEMM (B given as B^T, NxK)
// One wave computes a 32x32 output tile (2x2 WMMA register tiling)
__global__ void k_wmma_gemm(const unsigned short* __restrict__ A, int lda,
                            const unsigned short* __restrict__ Btr, int ldb,
                            const float* __restrict__ bias,
                            float* __restrict__ Cf, int ldc,
                            unsigned short* __restrict__ Cb, int ldcb, int K) {
    int lane = threadIdx.x;
    int n0 = blockIdx.x * 32;
    int m0 = blockIdx.y * 32;
    int i  = lane & 15;
    int kh = lane >> 4;
    const unsigned short* arow0 = A + (size_t)(m0 + i) * lda;
    const unsigned short* arow1 = A + (size_t)(m0 + 16 + i) * lda;
    const unsigned short* brow0 = Btr + (size_t)(n0 + i) * ldb;
    const unsigned short* brow1 = Btr + (size_t)(n0 + 16 + i) * ldb;

    v8f acc00 = {}, acc01 = {}, acc10 = {}, acc11 = {};
    for (int kb = 0; kb < K; kb += 32) {
        Frag fa0, fa1, fb0, fb1;
        load_fragA(fa0, arow0 + kb, kh);
        load_fragA(fa1, arow1 + kb, kh);
        load_fragB(fb0, brow0 + kb, kh);
        load_fragB(fb1, brow1 + kb, kh);
        __builtin_prefetch(brow0 + kb + 128, 0, 1);  // global_prefetch_b8
        acc00 = WMMA_BF16(fa0, fb0, acc00);
        acc01 = WMMA_BF16(fa0, fb1, acc01);
        acc10 = WMMA_BF16(fa1, fb0, acc10);
        acc11 = WMMA_BF16(fa1, fb1, acc11);
    }
    int r0 = m0 + kh * 8, r1 = m0 + 16 + kh * 8;
    int c0 = n0 + i,      c1 = n0 + 16 + i;
    float bv0 = bias ? bias[c0] : 0.0f;
    float bv1 = bias ? bias[c1] : 0.0f;
    for (int v = 0; v < 8; ++v) {
        float v00 = acc00[v] + bv0, v01 = acc01[v] + bv1;
        float v10 = acc10[v] + bv0, v11 = acc11[v] + bv1;
        if (Cf) {
            Cf[(size_t)(r0 + v) * ldc + c0] = v00;
            Cf[(size_t)(r0 + v) * ldc + c1] = v01;
            Cf[(size_t)(r1 + v) * ldc + c0] = v10;
            Cf[(size_t)(r1 + v) * ldc + c1] = v11;
        }
        if (Cb) {
            Cb[(size_t)(r0 + v) * ldcb + c0] = f2bf(v00);
            Cb[(size_t)(r0 + v) * ldcb + c1] = f2bf(v01);
            Cb[(size_t)(r1 + v) * ldcb + c0] = f2bf(v10);
            Cb[(size_t)(r1 + v) * ldcb + c1] = f2bf(v11);
        }
    }
}

// ------------------------------------------------- attention (one block per batch row)
__global__ void k_attention(const unsigned short* __restrict__ hidden, const float* __restrict__ ctrl,
                            const float* __restrict__ seq_mask, unsigned short* __restrict__ catbuf,
                            float* __restrict__ fmsg, int step) {
    int b = blockIdx.x, t = threadIdx.x;               // 256 threads
    __shared__ float att[TT];
    __shared__ float red[TT];
    const unsigned short* hrow = hidden + ((size_t)b * TT + t) * HSZ;
    const float* cr = ctrl + b * HSZ;
    float s = 0.0f;
    for (int h = 0; h < HSZ; ++h) s += bf2f(hrow[h]) * cr[h];
    s -= 1e30f * (1.0f - seq_mask[b * TT + t]);
    red[t] = s; __syncthreads();
    for (int o = 128; o > 0; o >>= 1) { if (t < o) red[t] = fmaxf(red[t], red[t + o]); __syncthreads(); }
    float mx = red[0]; __syncthreads();
    float e = __expf(s - mx);
    att[t] = e; red[t] = e; __syncthreads();
    for (int o = 128; o > 0; o >>= 1) { if (t < o) red[t] += red[t + o]; __syncthreads(); }
    float inv = 1.0f / red[0];
    __syncthreads();
    for (int h0 = 0; h0 < HSZ; h0 += TT) {
        int h = h0 + t;
        float acc = 0.0f;
        for (int tt = 0; tt < TT; ++tt)
            acc += att[tt] * bf2f(hidden[((size_t)b * TT + tt) * HSZ + h]);
        acc *= inv;
        fmsg[b * 3 * HSZ + step * HSZ + h] = acc;
        catbuf[b * 2048 + h] = f2bf(acc);
    }
}

// ------------------------------------------------- landmark embedding sum
__global__ void k_lmk_embed(const int* __restrict__ landmarks, const float* __restrict__ map_emb,
                            float* __restrict__ l_f, unsigned short* __restrict__ l_bf) {
    int blk = blockIdx.x;                 // b*16 + p
    int b = blk >> 4, p = blk & 15;
    const int* lm = landmarks + ((size_t)b * 16 + p) * 6;
    int i0 = lm[0], i1 = lm[1], i2 = lm[2], i3 = lm[3], i4 = lm[4], i5 = lm[5];
    for (int c = threadIdx.x; c < HSZ; c += blockDim.x) {
        float s = map_emb[i0 * HSZ + c] + map_emb[i1 * HSZ + c] + map_emb[i2 * HSZ + c]
                + map_emb[i3 * HSZ + c] + map_emb[i4 * HSZ + c] + map_emb[i5 * HSZ + c];
        l_f[(size_t)blk * HSZ + c] = s;
        l_bf[(size_t)blk * HSZ + c] = f2bf(s);
    }
}

// ------------------------------------------------- im2col for 4-tap cross conv on 4x4 grid
// Async-copy staging: global -> LDS via global_load_async_to_lds_b128 (ASYNCcnt),
// zero-fill OOB taps in LDS, then coalesced store of the gathered row to global.
__global__ void k_im2col(const unsigned short* __restrict__ in, unsigned short* __restrict__ out) {
    __shared__ unsigned short sh[4 * HSZ];              // 8 KB staging tile
    int blk = blockIdx.x;                               // b*16 + p
    int b = blk >> 4, p = blk & 15;
    int y = p >> 2, x = p & 3;
    const int dyv[4] = {-1, 0, 0, 1};
    const int dxv[4] = {0, -1, 1, 0};
    int tid = threadIdx.x;                              // 256 threads
    for (int k = 0; k < 2; ++k) {
        int ci = tid + k * 256;                         // 16B chunk id, 0..511
        int d  = ci >> 7;                               // tap
        int ch = ci & 127;                              // chunk within tap
        int ny = y + dyv[d], nx = x + dxv[d];
        bool ok = ((unsigned)ny < 4u) && ((unsigned)nx < 4u);
        unsigned short* ldsp = &sh[d * HSZ + ch * 8];
        if (ok) {
            unsigned lds_off = (unsigned)(uintptr_t)ldsp;
            const unsigned short* src = in + ((size_t)b * 16 + ny * 4 + nx) * HSZ + ch * 8;
            asm volatile("global_load_async_to_lds_b128 %0, %1, off"
                         :: "v"(lds_off), "v"(src) : "memory");
        } else {
            uint4 z = {0u, 0u, 0u, 0u};
            *(uint4*)ldsp = z;
        }
    }
    asm volatile("s_wait_asynccnt 0x0" ::: "memory");
    __syncthreads();
    for (int k = 0; k < 2; ++k) {
        int ci = tid + k * 256;
        uint4 v = *(const uint4*)&sh[ci * 8];
        *(uint4*)(out + (size_t)blk * 4096 + ci * 8) = v;
    }
}

// ------------------------------------------------- logits[b][n] = sum_c lmk * feature_msg
__global__ void k_logits(const float* __restrict__ l_f, const float* __restrict__ fmsg,
                         float* __restrict__ logits) {
    int n = blockIdx.x, b = blockIdx.y;
    __shared__ float red[256];
    float acc = 0.0f;
    for (int c = threadIdx.x; c < 3 * HSZ; c += 256) {
        int layer = c >> 10, cc = c & 1023;
        acc += l_f[((size_t)layer * BB * 16 + b * 16 + n) * HSZ + cc] * fmsg[b * 3 * HSZ + c];
    }
    red[threadIdx.x] = acc; __syncthreads();
    for (int o = 128; o > 0; o >>= 1) { if (threadIdx.x < o) red[threadIdx.x] += red[threadIdx.x + o]; __syncthreads(); }
    if (threadIdx.x == 0) logits[b * 16 + n] = red[0];
}

// ------------------------------------------------- final: softmax -> log_softmax(prob) -> loss/acc
__global__ void k_final(const float* __restrict__ logits, const int* __restrict__ ys,
                        float* __restrict__ out) {
    int b = threadIdx.x;                 // 128 threads, 1 block
    __shared__ float sl[BB], sa[BB];
    float l[16];
    float mx = -1e30f;
    for (int i = 0; i < 16; ++i) { l[i] = logits[b * 16 + i]; mx = fmaxf(mx, l[i]); }
    float sum = 0.0f;
    for (int i = 0; i < 16; ++i) { l[i] = __expf(l[i] - mx); sum += l[i]; }
    float inv = 1.0f / sum;
    for (int i = 0; i < 16; ++i) l[i] *= inv;            // prob
    float mx2 = -1e30f;
    for (int i = 0; i < 16; ++i) mx2 = fmaxf(mx2, l[i]);
    float s2 = 0.0f;
    for (int i = 0; i < 16; ++i) s2 += __expf(l[i] - mx2);
    float lse = mx2 + __logf(s2);
    int y = ys[b * 2] * 4 + ys[b * 2 + 1];
    float loss = -(l[y] - lse);
    int am = 0; float bv = l[0];
    for (int i = 1; i < 16; ++i) if (l[i] > bv) { bv = l[i]; am = i; }
    sl[b] = loss; sa[b] = (am == y) ? 1.0f : 0.0f;
    __syncthreads();
    for (int o = 64; o > 0; o >>= 1) { if (b < o) { sl[b] += sl[b + o]; sa[b] += sa[b + o]; } __syncthreads(); }
    if (b == 0) { out[0] = sl[0] / (float)BB; out[1] = sa[0] / (float)BB; }
}

// ================================================================ host launcher
extern "C" void kernel_launch(void* const* d_in, const int* in_sizes, int n_in,
                              void* d_out, int out_size, void* d_ws, size_t ws_size,
                              hipStream_t stream) {
    const int*   Xs        = (const int*)d_in[0];
    const float* seq_mask  = (const float*)d_in[1];
    const int*   landmarks = (const int*)d_in[2];
    const int*   ys        = (const int*)d_in[3];
    const float* emb       = (const float*)d_in[4];
    const float* W_ih_f    = (const float*)d_in[5];
    const float* W_hh_f    = (const float*)d_in[6];
    const float* b_f       = (const float*)d_in[7];
    const float* W_ih_b    = (const float*)d_in[8];
    const float* W_hh_b    = (const float*)d_in[9];
    const float* b_b       = (const float*)d_in[10];
    const float* feat_ctrl = (const float*)d_in[11];
    const float* upd_W     = (const float*)d_in[12];
    const float* upd_b     = (const float*)d_in[13];
    const float* map_emb   = (const float*)d_in[14];
    const float* conv_w    = (const float*)d_in[15];
    float* out = (float*)d_out;

    char* wsp = (char*)d_ws;
    size_t off = 0;
    auto carve = [&](size_t bytes) -> void* {
        off = (off + 255) & ~(size_t)255;
        void* p = wsp + off;
        off += bytes;
        return p;
    };
    unsigned short* emb_bf   = (unsigned short*)carve((size_t)2048 * 512 * 2);
    unsigned short* Wcat     = (unsigned short*)carve((size_t)2 * 2048 * 1024 * 2);
    unsigned short* updW_bf  = (unsigned short*)carve((size_t)1024 * 2048 * 2);
    unsigned short* convw_bf = (unsigned short*)carve((size_t)1024 * 4096 * 2);
    unsigned short* hbf      = (unsigned short*)carve((size_t)2 * BB * HDM * 2);
    float*          c_state  = (float*)carve((size_t)2 * BB * HDM * 4);
    float*          gates    = (float*)carve((size_t)2 * BB * NG * 4);
    unsigned short* hidden   = (unsigned short*)carve((size_t)BB * TT * HSZ * 2);
    float*          ctrl     = (float*)carve((size_t)BB * HSZ * 4);
    unsigned short* catbuf   = (unsigned short*)carve((size_t)BB * 2048 * 2);
    float*          fmsg     = (float*)carve((size_t)BB * 3 * HSZ * 4);
    float*          l_f      = (float*)carve((size_t)3 * BB * 16 * HSZ * 4);
    unsigned short* l0_bf    = (unsigned short*)carve((size_t)BB * 16 * HSZ * 2);
    unsigned short* l1_bf    = (unsigned short*)carve((size_t)BB * 16 * HSZ * 2);
    unsigned short* cat4     = (unsigned short*)carve((size_t)BB * 16 * 4096 * 2);
    float*          logits   = (float*)carve((size_t)BB * 16 * 4);

    // ---- prep: weight / state conversion
    k_f32_to_bf16<<<(2048 * 512) / 256, 256, 0, stream>>>(emb, emb_bf, 2048 * 512);
    k_f32_to_bf16<<<(1024 * 2048) / 256, 256, 0, stream>>>(upd_W, updW_bf, 1024 * 2048);
    k_build_wcat<<<(2 * 2048 * 1024) / 256, 256, 0, stream>>>(W_ih_f, W_hh_f, W_ih_b, W_hh_b, Wcat);
    k_build_convw<<<(1024 * 4096) / 256, 256, 0, stream>>>(conv_w, convw_bf);
    k_init_state<<<(2 * BB * HDM) / 256, 256, 0, stream>>>(c_state, hbf);
    k_ctrl_init<<<(BB * HSZ) / 256, 256, 0, stream>>>(feat_ctrl, ctrl, catbuf);
    k_lmk_embed<<<BB * 16, 256, 0, stream>>>(landmarks, map_emb, l_f, l0_bf);

    // ---- bidirectional LSTM: 256 sequential steps, both directions per launch
    for (int step = 0; step < TT; ++step) {
        k_lstm_step_gemm<<<dim3(NG / 32, BB / 32, 2), 32, 0, stream>>>(
            Xs, emb_bf, hbf, Wcat, b_f, b_b, gates, step);
        k_lstm_cell<<<(2 * BB * HDM) / 256, 256, 0, stream>>>(gates, c_state, hbf, hidden, step);
    }

    // ---- attention controller (3 steps; update GEMM for first two)
    for (int s = 0; s < 3; ++s) {
        k_attention<<<BB, TT, 0, stream>>>(hidden, ctrl, seq_mask, catbuf, fmsg, s);
        if (s < 2)
            k_wmma_gemm<<<dim3(HSZ / 32, BB / 32), 32, 0, stream>>>(
                catbuf, 2048, updW_bf, 2048, upd_b, ctrl, HSZ, catbuf + HSZ, 2048, 2048);
    }

    // ---- masked cross-conv stack as im2col GEMMs
    k_im2col<<<BB * 16, 256, 0, stream>>>(l0_bf, cat4);
    k_wmma_gemm<<<dim3(HSZ / 32, (BB * 16) / 32), 32, 0, stream>>>(
        cat4, 4096, convw_bf, 4096, nullptr,
        l_f + (size_t)1 * BB * 16 * HSZ, HSZ, l1_bf, HSZ, 4096);
    k_im2col<<<BB * 16, 256, 0, stream>>>(l1_bf, cat4);
    k_wmma_gemm<<<dim3(HSZ / 32, (BB * 16) / 32), 32, 0, stream>>>(
        cat4, 4096, convw_bf, 4096, nullptr,
        l_f + (size_t)2 * BB * 16 * HSZ, HSZ, nullptr, HSZ, 4096);

    // ---- logits + loss/acc
    k_logits<<<dim3(16, BB), 256, 0, stream>>>(l_f, fmsg, logits);
    k_final<<<1, BB, 0, stream>>>(logits, ys, out);

    (void)in_sizes; (void)n_in; (void)out_size; (void)ws_size;
}
